// GenericClassNet_23527830847711
// MI455X (gfx1250) — compile-verified, matrix operand order checked by reading, not run on hardware
//
#include <hip/hip_runtime.h>

// ---------------------------------------------------------------------------
// GIN network forward for MI455X (gfx1250, wave32, WMMA).
// Strategy: every Linear is a fused kernel
//     out = [relu(BN_affine(in))] @ W_f16 + b        (fp32 accum via v_wmma)
// where BN_affine (scale/shift per column) comes from a two-kernel column
// reduction of the *previous* layer's raw linear output. This keeps every
// 128->128 layer at ~2x51MB of HBM traffic (the minimum) on a 23.3 TB/s part.
// Weights are packed once per launch into the ISA's per-lane B-fragment
// layout so the WMMA hot loop is contiguous 32B loads.
// ---------------------------------------------------------------------------

typedef __attribute__((ext_vector_type(16))) _Float16 v16h;
typedef __attribute__((ext_vector_type(8)))  float    v8f;
typedef __attribute__((ext_vector_type(4)))  float    v4f;

// ---------------------------------------------------------------------------
// Pack fp32 weight [K=128, dout] (row-major) into WMMA B-fragment order:
// frag(kc, ntile, lane) = 16 contiguous halves per lane.
// Half hi of lane L in k-chunk kc maps to  k = kc*32 + (hi>>3)*16 + (hi&7) + ((L>=16)?8:0)
//                                          n = ntile*16 + (L&15)
// (mirrors the 16-bit A-matrix VGPR table in CDNA5 ISA §7.12.2)
// ---------------------------------------------------------------------------
__global__ void pack_w_kernel(const float* __restrict__ W,
                              _Float16* __restrict__ P, int dout) {
  int idx = blockIdx.x * blockDim.x + threadIdx.x;
  int total = 128 * dout;
  if (idx >= total) return;
  int hi   = idx & 15;
  int f    = idx >> 4;
  int lane = f & 31;
  int rest = f >> 5;
  int ntiles = dout >> 4;
  int nt = rest % ntiles;
  int kc = rest / ntiles;
  int k = kc * 32 + ((hi >> 3) << 4) + (hi & 7) + ((lane >> 4) << 3);
  int n = nt * 16 + (lane & 15);
  P[idx] = (_Float16)W[k * dout + n];
}

// ---------------------------------------------------------------------------
// Fused GEMM:  C[M, dout(ldc)] = act(A[M,128]) @ W + bias
//   act(x) = aff ? relu(x*scale[k] + shift[k]) : x   (per input column k)
// Block = 256 threads = 8 waves; each wave computes a 16-row strip,
// NT 16x16 C tiles, K-loop of 4 x v_wmma_f32_16x16x32_f16.
// ---------------------------------------------------------------------------
template <int NT>
__global__ __launch_bounds__(256) void gemm_wmma_kernel(
    const float* __restrict__ A, const _Float16* __restrict__ P,
    const float* __restrict__ bias, const float* __restrict__ aff,
    float* __restrict__ C, int M, int ldc) {
  const int lane = threadIdx.x & 31;
  const int wave = threadIdx.x >> 5;
  const int tilebase = blockIdx.x * 128 + wave * 16;
  const int hiK = (lane >> 4) << 3;  // +8 for lanes 16..31

  int arow = tilebase + (lane & 15);
  if (arow > M - 1) arow = M - 1;  // clamp (stores are guarded)
  const float* __restrict__ Arow = A + (size_t)arow * 128;

  v8f acc[NT] = {};

#pragma unroll
  for (int kc = 0; kc < 4; ++kc) {
    const int k0 = kc * 32 + hiK;
    v4f x0 = *(const v4f*)(Arow + k0);
    v4f x1 = *(const v4f*)(Arow + k0 + 4);
    v4f x2 = *(const v4f*)(Arow + k0 + 16);
    v4f x3 = *(const v4f*)(Arow + k0 + 20);
    if (aff) {
      const float* sh = aff + 128;
      v4f s0 = *(const v4f*)(aff + k0);
      v4f s1 = *(const v4f*)(aff + k0 + 4);
      v4f s2 = *(const v4f*)(aff + k0 + 16);
      v4f s3 = *(const v4f*)(aff + k0 + 20);
      v4f t0 = *(const v4f*)(sh + k0);
      v4f t1 = *(const v4f*)(sh + k0 + 4);
      v4f t2 = *(const v4f*)(sh + k0 + 16);
      v4f t3 = *(const v4f*)(sh + k0 + 20);
#pragma unroll
      for (int j = 0; j < 4; ++j) {
        x0[j] = fmaxf(fmaf(x0[j], s0[j], t0[j]), 0.0f);
        x1[j] = fmaxf(fmaf(x1[j], s1[j], t1[j]), 0.0f);
        x2[j] = fmaxf(fmaf(x2[j], s2[j], t2[j]), 0.0f);
        x3[j] = fmaxf(fmaf(x3[j], s3[j], t3[j]), 0.0f);
      }
    }
    v16h a;
#pragma unroll
    for (int j = 0; j < 4; ++j) {
      a[j]      = (_Float16)x0[j];
      a[4 + j]  = (_Float16)x1[j];
      a[8 + j]  = (_Float16)x2[j];
      a[12 + j] = (_Float16)x3[j];
    }
    const v16h* __restrict__ Bp = (const v16h*)P + (size_t)(kc * NT) * 32 + lane;
#pragma unroll
    for (int nt = 0; nt < NT; ++nt) {
      v16h b = Bp[(size_t)nt * 32];
      acc[nt] = __builtin_amdgcn_wmma_f32_16x16x32_f16(
          false, a, false, b, (short)0, acc[nt], false, false);
    }
  }

  // C/D layout: lane 0-15 -> M=r, lane 16-31 -> M=8+r, N = (lane&15)+16*nt
  const int mbase = tilebase + hiK;
#pragma unroll
  for (int nt = 0; nt < NT; ++nt) {
    const int n = nt * 16 + (lane & 15);
    const float bb = bias ? bias[n] : 0.0f;
#pragma unroll
    for (int r = 0; r < 8; ++r) {
      const int m = mbase + r;
      if (m < M) C[(size_t)m * ldc + n] = acc[nt][r] + bb;
    }
  }
}

// ---------------------------------------------------------------------------
// Column stats (sum, sumsq) of optionally relu(affine(x)) over rows.
// 128 threads/block, each block reduces a row chunk, then atomics.
// ---------------------------------------------------------------------------
__global__ void colstats_kernel(const float* __restrict__ A, int M,
                                const float* __restrict__ preaff,
                                float* __restrict__ sum,
                                float* __restrict__ sumsq, int rowsPerBlock) {
  int c = threadIdx.x;  // column 0..127
  float s = 1.0f, t = 0.0f;
  if (preaff) { s = preaff[c]; t = preaff[128 + c]; }
  int r0 = blockIdx.x * rowsPerBlock;
  int r1 = r0 + rowsPerBlock;
  if (r1 > M) r1 = M;
  float a = 0.0f, q = 0.0f;
  for (int r = r0; r < r1; ++r) {
    float v = A[(size_t)r * 128 + c];
    if (preaff) v = fmaxf(fmaf(v, s, t), 0.0f);
    a += v;
    q += v * v;
  }
  atomicAdd(&sum[c], a);
  atomicAdd(&sumsq[c], q);
}

// Fold stats + gamma/beta into (scale, shift); re-zero accumulators for reuse.
__global__ void bn_finalize_kernel(float* __restrict__ sum,
                                   float* __restrict__ sumsq,
                                   const float* __restrict__ g,
                                   const float* __restrict__ bt, float invM,
                                   float* __restrict__ affout) {
  int c = threadIdx.x;
  float mu = sum[c] * invM;
  float var = sumsq[c] * invM - mu * mu;
  float rs = rsqrtf(var + 1e-5f);
  float sc = g[c] * rs;
  affout[c] = sc;
  affout[128 + c] = bt[c] - mu * sc;
  sum[c] = 0.0f;
  sumsq[c] = 0.0f;
}

__global__ void zero_kernel(float* __restrict__ p, size_t n) {
  size_t i = blockIdx.x * (size_t)blockDim.x + threadIdx.x;
  if (i < n) p[i] = 0.0f;
}

// T = (1 + eps) * H   (eps is a device scalar leaf)
__global__ void scale_init_kernel(const float* __restrict__ H,
                                  float* __restrict__ T,
                                  const float* __restrict__ epsp, size_t n) {
  size_t i = blockIdx.x * (size_t)blockDim.x + threadIdx.x;
  if (i >= n) return;
  float c = 1.0f + *epsp;
  T[i] = c * H[i];
}

// neigh accumulation: one wave per edge, 4 features per lane, fp32 atomics.
__global__ void edge_scatter_kernel(const float* __restrict__ H,
                                    const int* __restrict__ src,
                                    const int* __restrict__ dst,
                                    float* __restrict__ T, int E) {
  int e = (int)((blockIdx.x * (size_t)blockDim.x + threadIdx.x) >> 5);
  int lane = threadIdx.x & 31;
  if (e >= E) return;
  int s = src[e], d = dst[e];
  v4f v = *(const v4f*)(H + (size_t)s * 128 + lane * 4);
  float* o = T + (size_t)d * 128 + lane * 4;
  atomicAdd(o + 0, v[0]);
  atomicAdd(o + 1, v[1]);
  atomicAdd(o + 2, v[2]);
  atomicAdd(o + 3, v[3]);
}

// H += relu(affB(relu(affA(T))))
__global__ void gin_residual_kernel(float* __restrict__ H,
                                    const float* __restrict__ T,
                                    const float* __restrict__ affA,
                                    const float* __restrict__ affB, size_t n) {
  size_t i = blockIdx.x * (size_t)blockDim.x + threadIdx.x;
  if (i >= n) return;
  int c = (int)(i & 127);
  float v = T[i];
  v = fmaxf(fmaf(v, affA[c], affA[128 + c]), 0.0f);
  v = fmaxf(fmaf(v, affB[c], affB[128 + c]), 0.0f);
  H[i] += v;
}

// mean pooling over query graphs: scatter sums + counts, then divide
__global__ void pool_scatter_kernel(const float* __restrict__ Hq,
                                    const int* __restrict__ gid,
                                    float* __restrict__ qrep,
                                    float* __restrict__ qcnt, int NQn) {
  int i = blockIdx.x * blockDim.x + threadIdx.x;
  if (i >= NQn * 64) return;
  int nrow = i >> 6, c = i & 63;
  int g = gid[nrow];
  atomicAdd(&qrep[g * 64 + c], Hq[(size_t)nrow * 64 + c]);
  if (c == 0) atomicAdd(&qcnt[g], 1.0f);
}

__global__ void pool_div_kernel(float* __restrict__ qrep,
                                const float* __restrict__ qcnt) {
  int i = blockIdx.x * blockDim.x + threadIdx.x;
  if (i >= 512 * 64) return;
  qrep[i] = qrep[i] / fmaxf(qcnt[i >> 6], 1.0f);
}

// fill concat buffer columns 64..127 with q_rep[node_gid[m]]
__global__ void concat_gather_kernel(float* __restrict__ Cg,
                                     const float* __restrict__ qrep,
                                     const int* __restrict__ gid, int Nn) {
  int i = blockIdx.x * blockDim.x + threadIdx.x;
  if (i >= Nn * 64) return;
  int m = i >> 6, c = i & 63;
  Cg[(size_t)m * 128 + 64 + c] = qrep[gid[m] * 64 + c];
}

// final predict layer: out[m] = sigmoid( relu(aff(x[m,:])) . W + b ), wave/row
__global__ void predict_final_kernel(const float* __restrict__ X,
                                     const float* __restrict__ aff,
                                     const float* __restrict__ W,
                                     const float* __restrict__ b,
                                     float* __restrict__ out, int Mn) {
  int gtid = blockIdx.x * blockDim.x + threadIdx.x;
  int row = gtid >> 5;
  int lane = threadIdx.x & 31;
  if (row >= Mn) return;
  const float* x = X + (size_t)row * 128 + lane * 4;
  float acc = 0.0f;
#pragma unroll
  for (int j = 0; j < 4; ++j) {
    int k = lane * 4 + j;
    float v = fmaxf(fmaf(x[j], aff[k], aff[128 + k]), 0.0f);
    acc += v * W[k];
  }
#pragma unroll
  for (int off = 16; off > 0; off >>= 1) acc += __shfl_xor(acc, off, 32);
  if (lane == 0) out[row] = 1.0f / (1.0f + __expf(-(acc + b[0])));
}

// ---------------------------------------------------------------------------
// Host orchestration
// ---------------------------------------------------------------------------
extern "C" void kernel_launch(void* const* d_in, const int* in_sizes, int n_in,
                              void* d_out, int out_size, void* d_ws,
                              size_t ws_size, hipStream_t stream) {
  (void)in_sizes; (void)out_size;
  const int N = 100000, NQ = 20000, EG = 800000, EQ = 320000;
  if (n_in < 244) return;  // expects insertion-order pytree flatten (244 leaves)

  auto F = [&](int i) -> const float* { return (const float*)d_in[i]; };
  auto I = [&](int i) -> const int* { return (const int*)d_in[i]; };

  // --- leaf index map (insertion-order flatten of setup_inputs()) ---
  const int IDX_EMB_W = 0, IDX_EMB_B = 1;
  auto gin_g_base = [](int i) { return 2 + 13 * i; };    // eps,W0,b0,g0,bt0,W1,b1,g1,bt1,gA,btA,gN,btN
  auto gin_q_base = [](int i) { return 54 + 13 * i; };
  auto mlp_g_base = [](int j) { return 106 + 4 * j; };   // W,b,g,bt
  auto mlp_q_base = [](int j) { return 138 + 4 * j; };
  auto pred_base  = [](int j) { return 170 + 4 * j; };
  const int IDX_X = 234, IDX_SRC = 236, IDX_DST = 237, IDX_GID = 238,
            IDX_XQ = 239, IDX_SRCQ = 241, IDX_DSTQ = 242, IDX_GIDQ = 243;

  // --- workspace carve ---
  char* wsb = (char*)d_ws;
  size_t off = 0;
  auto carve = [&](size_t bytes) -> char* {
    char* p = wsb + off;
    off = (off + bytes + 255) & ~(size_t)255;
    return p;
  };
  float* HG = (float*)carve((size_t)N * 128 * 4);   // residual h (g), later concat buf
  float* T1 = (float*)carve((size_t)N * 128 * 4);
  float* T2 = (float*)carve((size_t)N * 128 * 4);
  float* HQ = (float*)carve((size_t)NQ * 128 * 4);
  float* U1 = (float*)carve((size_t)NQ * 128 * 4);
  float* U2 = (float*)carve((size_t)NQ * 128 * 4);
  float* qrep = (float*)carve((size_t)(512 * 64 + 512) * 4);  // +counts
  float* qcnt = qrep + 512 * 64;
  float* cstat = (float*)carve(256 * 4);  // colsum[128] | colsumsq[128]
  float* csum = cstat, *csq = cstat + 128;
  float* affs = (float*)carve((size_t)64 * 256 * 4);  // 64 (scale,shift) slots
  _Float16* packbuf = (_Float16*)carve((size_t)48 * 128 * 128 * 2);
  if (off > ws_size) return;  // workspace too small; nothing safe to do

  // --- zero scratch that is accumulated into ---
  zero_kernel<<<1, 256, 0, stream>>>(cstat, 256);
  zero_kernel<<<(512 * 65 + 255) / 256, 256, 0, stream>>>(qrep, 512 * 65);

  // --- pack all GEMM weights to f16 fragment layout ---
  int wleaf[48], wdout[48], nw = 0;
  wleaf[nw] = IDX_EMB_W; wdout[nw++] = 128;
  for (int i = 0; i < 4; ++i) {
    int b = gin_g_base(i);
    wleaf[nw] = b + 1; wdout[nw++] = 128;
    wleaf[nw] = b + 5; wdout[nw++] = 128;
  }
  for (int i = 0; i < 4; ++i) {
    int b = gin_q_base(i);
    wleaf[nw] = b + 1; wdout[nw++] = 128;
    wleaf[nw] = b + 5; wdout[nw++] = 128;
  }
  for (int j = 0; j < 8; ++j) { wleaf[nw] = mlp_g_base(j); wdout[nw++] = (j == 7) ? 64 : 128; }
  for (int j = 0; j < 8; ++j) { wleaf[nw] = mlp_q_base(j); wdout[nw++] = (j == 7) ? 64 : 128; }
  for (int j = 0; j < 15; ++j) { wleaf[nw] = pred_base(j); wdout[nw++] = 128; }

  _Float16* ppack[48];
  size_t poff = 0;
  for (int w = 0; w < nw; ++w) {
    ppack[w] = packbuf + poff;
    int total = 128 * wdout[w];
    pack_w_kernel<<<(total + 255) / 256, 256, 0, stream>>>(F(wleaf[w]), ppack[w], wdout[w]);
    poff += (size_t)total;
  }
  auto packOf = [&](int leaf) -> _Float16* {
    for (int w = 0; w < nw; ++w)
      if (wleaf[w] == leaf) return ppack[w];
    return nullptr;
  };

  // --- helpers ---
  auto gemm = [&](const float* Ain, int leafW, const float* bias,
                  const float* aff, float* C, int M, int dout, int ldc) {
    _Float16* P = packOf(leafW);
    unsigned grid = (unsigned)((M + 127) / 128);
    if (dout == 128)
      gemm_wmma_kernel<8><<<grid, 256, 0, stream>>>(Ain, P, bias, aff, C, M, ldc);
    else
      gemm_wmma_kernel<4><<<grid, 256, 0, stream>>>(Ain, P, bias, aff, C, M, ldc);
  };
  int slot = 0;
  auto bnstats = [&](const float* Ain, int M, const float* preaff,
                     const float* g, const float* bt) -> float* {
    const int rows = 512;
    unsigned grid = (unsigned)((M + rows - 1) / rows);
    colstats_kernel<<<grid, 128, 0, stream>>>(Ain, M, preaff, csum, csq, rows);
    float* o = affs + (size_t)slot * 256;
    ++slot;
    bn_finalize_kernel<<<1, 128, 0, stream>>>(csum, csq, g, bt, 1.0f / (float)M, o);
    return o;
  };
  auto gin_layer = [&](int base, float* H, float* Ta, float* Tb, int M,
                       const int* srcp, const int* dstp, int E) {
    size_t n = (size_t)M * 128;
    unsigned eg = (unsigned)((n + 255) / 256);
    scale_init_kernel<<<eg, 256, 0, stream>>>(H, Ta, F(base + 0), n);
    edge_scatter_kernel<<<(unsigned)((E + 7) / 8), 256, 0, stream>>>(H, srcp, dstp, Ta, E);
    gemm(Ta, base + 1, F(base + 2), nullptr, Tb, M, 128, 128);           // mlp[0]
    float* a0 = bnstats(Tb, M, nullptr, F(base + 3), F(base + 4));
    gemm(Tb, base + 5, F(base + 6), a0, Ta, M, 128, 128);                // mlp[1]
    float* aA = bnstats(Ta, M, nullptr, F(base + 9), F(base + 10));      // apply_bn
    float* aB = bnstats(Ta, M, aA, F(base + 11), F(base + 12));          // node_bn
    gin_residual_kernel<<<eg, 256, 0, stream>>>(H, Ta, aA, aB, n);
  };
  auto head_mlp = [&](auto baseFn, float* in0, float* Ta, float* Tb, int M,
                      float* outC, int out_ldc) {
    const float* cur = in0;
    const float* aff = nullptr;
    float* buf[2] = {Ta, Tb};
    int pb = 0;
    for (int j = 0; j < 7; ++j) {
      int b = baseFn(j);
      float* o = buf[pb]; pb ^= 1;
      gemm(cur, b, F(b + 1), aff, o, M, 128, 128);
      aff = bnstats(o, M, nullptr, F(b + 2), F(b + 3));
      cur = o;
    }
    int b = baseFn(7);
    gemm(cur, b, F(b + 1), aff, outC, M, 64, out_ldc);
  };

  // ---- query branch ----
  gemm(F(IDX_XQ), IDX_EMB_W, F(IDX_EMB_B), nullptr, HQ, NQ, 128, 128);
  for (int i = 0; i < 4; ++i)
    gin_layer(gin_q_base(i), HQ, U1, U2, NQ, I(IDX_SRCQ), I(IDX_DSTQ), EQ);
  head_mlp(mlp_q_base, HQ, U1, U2, NQ, HQ, 64);  // H_q [NQ,64] into HQ (ld=64)
  pool_scatter_kernel<<<(NQ * 64 + 255) / 256, 256, 0, stream>>>(HQ, I(IDX_GIDQ), qrep, qcnt, NQ);
  pool_div_kernel<<<(512 * 64 + 255) / 256, 256, 0, stream>>>(qrep, qcnt);

  // ---- graph branch ----
  gemm(F(IDX_X), IDX_EMB_W, F(IDX_EMB_B), nullptr, HG, N, 128, 128);
  for (int i = 0; i < 4; ++i)
    gin_layer(gin_g_base(i), HG, T1, T2, N, I(IDX_SRC), I(IDX_DST), EG);
  // head writes H_g into concat buffer (HG) columns 0..63; h is dead by then
  head_mlp(mlp_g_base, HG, T1, T2, N, HG, 128);
  concat_gather_kernel<<<(unsigned)(((size_t)N * 64 + 255) / 256), 256, 0, stream>>>(
      HG, qrep, I(IDX_GID), N);

  // ---- predict MLP (15 x 128->128 with BN+ReLU, then 128->1 + sigmoid) ----
  {
    const float* cur = HG;
    const float* aff = nullptr;
    float* buf[2] = {T1, T2};
    int pb = 0;
    for (int j = 0; j < 15; ++j) {
      int b = pred_base(j);
      float* o = buf[pb]; pb ^= 1;
      gemm(cur, b, F(b + 1), aff, o, N, 128, 128);
      aff = bnstats(o, N, nullptr, F(b + 2), F(b + 3));
      cur = o;
    }
    int b = pred_base(15);
    predict_final_kernel<<<(unsigned)(((size_t)N * 32 + 255) / 256), 256, 0, stream>>>(
        cur, aff, F(b), F(b + 1), (float*)d_out, N);
  }
}
